// MessagePassingNetwork_33809982554511
// MI455X (gfx1250) — matrix operand order repaired and checked by reading, not compiled, not run on hardware
//
#include <hip/hip_runtime.h>

// MI455X / gfx1250 implementation of the message-passing network.
// All GEMM-shaped work runs on v_wmma_f32_16x16x32_bf16 (bf16 A/B, f32 acc).
//
// Pipeline (all on `stream`):
//   0. memset msg buffers (covers the zero row/col of each direction)
//   1. pack_weights: pre-pack all conv weights into WMMA A-fragment layout (bf16)
//   2. transpose_k : features/data_term -> transposed copies for left/right dirs
//   3. msg_kernel  : 4 dirs x 4 batch x 255 lines; 65->64->64->1 1-D conv stack
//   4. agg_layer1  : im2col (45->64 padded K) 3x3 conv, 5->64, ReLU  (NHWC bf16)
//   5. agg_conv3x3<64,true> : 3x3 conv 64->64, ReLU                 (NHWC bf16)
//   6. agg_conv3x3<1,false> : 3x3 conv 64->1 + bias -> d_out (f32)
//
// v2: message-line LDS buffers carry one zero "guard" position at each end
//     ([-1] and [256]) so all B-fragment / data-term / layer3 reads are
//     unconditional ds loads (no EXEC save/restore or zero-splats around WMMA).

typedef __bf16 bf16;
typedef __bf16 v8bf  __attribute__((ext_vector_type(8)));
typedef __bf16 v16bf __attribute__((ext_vector_type(16)));
typedef float  v8f   __attribute__((ext_vector_type(8)));

#define DEV static __device__ __forceinline__

DEV v8f zero8() {
  v8f z;
#pragma unroll
  for (int i = 0; i < 8; ++i) z[i] = 0.0f;
  return z;
}

DEV v16bf load16(const bf16* p) {
  const v8bf lo = *(const v8bf*)(p);
  const v8bf hi = *(const v8bf*)(p + 8);
  return __builtin_shufflevector(lo, hi, 0, 1, 2, 3, 4, 5, 6, 7, 8, 9, 10, 11, 12, 13, 14, 15);
}

DEV v8f wmma_bf16(v16bf a, v16bf b, v8f c) {
  // D = A(16x32) * B(32x16) + C, f32 accumulate
  return __builtin_amdgcn_wmma_f32_16x16x32_bf16(false, a, false, b, (short)0, c, false, false);
}

// ---------------------------------------------------------------------------
// Geometry constants
// ---------------------------------------------------------------------------
#define NB   4
#define NC   32
#define NH   256
#define NW   256
#define HID  64
#define LSTR 72   // LDS channel stride (>=65, multiple of 8 -> 16B aligned frags)

// packed A-fragment table (each fragment: 32 lanes x 16 bf16 = 512 elems):
//   [  0, 48): msg layer1, idx = ((var*3+tap)*2+kt)*4 + m   (var: 0=row taps, 1=col taps)
//   [ 48, 96): msg layer2, same indexing
//   [ 96,104): agg layer1, idx = kt*4 + m                   (K = tap*5+c, 45 valid)
//   [104,176): agg layer2, idx = (tap*2+kt)*4 + m           (tap = dy*3+dx)
//   [176,194): agg layer3, idx = tap*2 + kt                 (M rows 1..15 zero)
#define FRAG_MSG_L1 0
#define FRAG_MSG_L2 48
#define FRAG_AGG1   96
#define FRAG_AGG2   104
#define FRAG_AGG3   176
#define N_FRAGS     194

// ---------------------------------------------------------------------------
// Weight packing into WMMA A fragments (16-bit A 16x32), this file's contiguous k order
//   lane<16 : M=lane,    K(e) = e
//   lane>=16: M=lane-16, K(e) = 16 + e
// ---------------------------------------------------------------------------
__global__ void pack_weights(const float* __restrict__ g1w, const float* __restrict__ g2w,
                             const float* __restrict__ a1w, const float* __restrict__ a2w,
                             const float* __restrict__ a3w, bf16* __restrict__ packed) {
  const int f = blockIdx.x;
  const int lane = threadIdx.x;  // 32 threads
  const int Mrow = lane & 15;
  for (int e = 0; e < 16; ++e) {
    const int v = e >> 1, j = e & 1;
    const int base = (lane < 16) ? 0 : 16;
    const int Kl = base + e;
    float val = 0.0f;
    if (f < FRAG_MSG_L2) {                       // msg layer1 (g1w [64][65][3][3])
      const int g = f, var = g / 24, rem = g % 24;
      const int tap = rem >> 3, kt = (rem >> 2) & 1, m = rem & 3;
      const int o = m * 16 + Mrow, K = kt * 32 + Kl;
      const int toff = var ? (tap * 3 + 1) : (3 + tap);
      val = g1w[(o * 65 + K) * 9 + toff];
    } else if (f < FRAG_AGG1) {                  // msg layer2 (g2w [64][64][3][3])
      const int g = f - FRAG_MSG_L2, var = g / 24, rem = g % 24;
      const int tap = rem >> 3, kt = (rem >> 2) & 1, m = rem & 3;
      const int o = m * 16 + Mrow, K = kt * 32 + Kl;
      const int toff = var ? (tap * 3 + 1) : (3 + tap);
      val = g2w[(o * 64 + K) * 9 + toff];
    } else if (f < FRAG_AGG2) {                  // agg layer1 (a1w [64][5][3][3])
      const int g = f - FRAG_AGG1, kt = g >> 2, m = g & 3;
      const int o = m * 16 + Mrow, K = kt * 32 + Kl;
      if (K < 45) {
        const int tap = K / 5, c = K - tap * 5, dy = tap / 3, dx = tap - dy * 3;
        val = a1w[(o * 5 + c) * 9 + dy * 3 + dx];
      }
    } else if (f < FRAG_AGG3) {                  // agg layer2 (a2w [64][64][3][3])
      const int g = f - FRAG_AGG2, m = g & 3, kt = (g >> 2) & 1, tap = g >> 3;
      const int dy = tap / 3, dx = tap - dy * 3;
      const int o = m * 16 + Mrow, K = kt * 32 + Kl;
      val = a2w[(o * 64 + K) * 9 + dy * 3 + dx];
    } else {                                     // agg layer3 (a3w [1][64][3][3])
      const int g = f - FRAG_AGG3, kt = g & 1, tap = g >> 1;
      const int dy = tap / 3, dx = tap - dy * 3;
      const int K = kt * 32 + Kl;
      if (Mrow == 0) val = a3w[K * 9 + dy * 3 + dx];
    }
    packed[(size_t)f * 512 + lane * 16 + e] = (bf16)val;
  }
}

// ---------------------------------------------------------------------------
// Transpose features (+ data_term as plane 32) for the left/right directions
// ---------------------------------------------------------------------------
__global__ void transpose_k(const float* __restrict__ feat, const float* __restrict__ dtin,
                            float* __restrict__ featT, float* __restrict__ dtT) {
  __shared__ float tile[32][33];
  const int plane = blockIdx.z;
  const int b = plane / 33, ch = plane % 33;
  const float* src = (ch < 32) ? feat + (size_t)(b * 32 + ch) * (NH * NW) : dtin + (size_t)b * (NH * NW);
  float* dst = (ch < 32) ? featT + (size_t)(b * 32 + ch) * (NH * NW) : dtT + (size_t)b * (NH * NW);
  const int x0 = blockIdx.x * 32, y0 = blockIdx.y * 32;
  const int tx = threadIdx.x, ty = threadIdx.y;
#pragma unroll
  for (int i = 0; i < 4; ++i)
    tile[ty + i * 8][tx] = src[(size_t)(y0 + ty + i * 8) * NW + x0 + tx];
  __syncthreads();
#pragma unroll
  for (int i = 0; i < 4; ++i)
    dst[(size_t)(x0 + ty + i * 8) * NH + y0 + tx] = tile[tx][ty + i * 8];
}

// ---------------------------------------------------------------------------
// One GEMM layer of the 1-D message stack (64 outputs x 256 positions)
//   src/dst: LDS line buffers with a zero guard position at [-1] and [256]
//   (pointers passed already offset so index -1 is valid & zero).
//   3 taps x 2 K-tiles of WMMA; all LDS reads unconditional.
// ---------------------------------------------------------------------------
template <bool HAS_DT, bool RELU>
DEV void msg_gemm_layer(const bf16* __restrict__ src, bf16* __restrict__ dst,
                        const bf16* __restrict__ afragBase, const float* __restrict__ biasS,
                        const float* __restrict__ wdtS, int mTile, int ntBase, int lane) {
  v16bf A[3][2];
#pragma unroll
  for (int tap = 0; tap < 3; ++tap)
#pragma unroll
    for (int kt = 0; kt < 2; ++kt)
      A[tap][kt] = load16(afragBase + (size_t)((tap * 2 + kt) * 4 + mTile) * 512 + lane * 16);

  const int n = lane & 15;
  const int hi16 = lane >> 4;
  const int chBase = mTile * 16 + hi16 * 8;

  for (int nt = ntBase; nt < ntBase + 8; ++nt) {
    v8f acc = zero8();
#pragma unroll
    for (int tap = 0; tap < 3; ++tap) {
      const int p = nt * 16 + n + tap - 1;          // in [-1, 256]: guards cover it
      const bf16* srcp = src + p * LSTR + hi16 * 16;
#pragma unroll
      for (int kt = 0; kt < 2; ++kt)
        acc = wmma_bf16(A[tap][kt], load16(srcp + kt * 32), acc);
    }
    const int pos = nt * 16 + n;
    float dval[3] = {0.f, 0.f, 0.f};
    if constexpr (HAS_DT) {
#pragma unroll
      for (int tap = 0; tap < 3; ++tap)
        dval[tap] = (float)src[(pos + tap - 1) * LSTR + 64];  // guard -> zero
    }
    v8bf o;
#pragma unroll
    for (int v = 0; v < 8; ++v) {
      const int ch = chBase + v;
      float x = acc[v] + biasS[ch];
      if constexpr (HAS_DT) {
#pragma unroll
        for (int tap = 0; tap < 3; ++tap) x += wdtS[ch * 3 + tap] * dval[tap];
      }
      if constexpr (RELU) x = fmaxf(x, 0.0f);
      o[v] = (bf16)x;
    }
    *(v8bf*)&dst[pos * LSTR + chBase] = o;
  }
}

// ---------------------------------------------------------------------------
// Message generation: one block = (line L, batch b, direction dir)
// dirs: 0=up 1=down (row-wise, conv along W) ; 2=left 3=right (transposed data)
// ---------------------------------------------------------------------------
__global__ __launch_bounds__(256) void msg_kernel(
    const float* __restrict__ feat, const float* __restrict__ dt,
    const float* __restrict__ featT, const float* __restrict__ dtT,
    const float* __restrict__ g1w, const float* __restrict__ g1b,
    const float* __restrict__ g2b, const float* __restrict__ g3w,
    const float* __restrict__ g3b, const bf16* __restrict__ packed,
    float* __restrict__ msg, const int* __restrict__ iters) {
  if (iters[0] <= 0) return;  // msgs pre-zeroed -> matches iterations==0 path

  // 258 positions: row 0 = guard (-1), rows 1..256 = positions 0..255, row 257 = guard (256)
  __shared__ bf16 rawA[(NW + 2) * LSTR];   // x (65ch) then h2
  __shared__ bf16 rawB[(NW + 2) * LSTR];   // h1
  __shared__ float wdts[192], w3s[192], b1s[64], b2s[64];
  bf16* bufA = rawA + LSTR;                // index -1 valid
  bf16* bufB = rawB + LSTR;

  const int L = blockIdx.x, b = blockIdx.y, dir = blockIdx.z;
  const bool tr = (dir >= 2);
  const int var = tr ? 1 : 0;
  const int nodeL = (dir == 0 || dir == 2) ? L + 1 : L;
  const int nbrL  = (dir == 0 || dir == 2) ? L     : L + 1;
  const float* fsrc = tr ? featT : feat;
  const float* dsrc = tr ? dtT : dt;
  const int tid = threadIdx.x;

  // zero the guard rows of both buffers (never overwritten: stores hit rows 1..256)
  if (tid < 2 * LSTR) {
    const int row = (tid < LSTR) ? 0 : (NW + 1);
    const int c = (tid < LSTR) ? tid : tid - LSTR;
    rawA[row * LSTR + c] = (bf16)0.0f;
    rawB[row * LSTR + c] = (bf16)0.0f;
  }
  // stage small params
  if (tid < 192) {
    const int ch = tid / 3, k = tid % 3;
    const int toff = var ? (k * 3 + 1) : (3 + k);
    w3s[tid]  = g3w[ch * 9 + toff];
    wdts[tid] = g1w[(ch * 65 + 64) * 9 + toff];  // data-term channel of layer1
  }
  if (tid < 64) { b1s[tid] = g1b[tid]; b2s[tid] = g2b[tid]; }

  // stage x: ch 0..31 node line, 32..63 neighbor line, 64 data_term (node line)
  {
    const int pos = tid;
    for (int ch = 0; ch < 32; ++ch)
      bufA[pos * LSTR + ch] = (bf16)fsrc[(size_t)(b * 32 + ch) * (NH * NW) + nodeL * NW + pos];
    for (int ch = 0; ch < 32; ++ch)
      bufA[pos * LSTR + 32 + ch] = (bf16)fsrc[(size_t)(b * 32 + ch) * (NH * NW) + nbrL * NW + pos];
    bufA[pos * LSTR + 64] = (bf16)dsrc[(size_t)b * (NH * NW) + nodeL * NW + pos];
  }
  __syncthreads();

  const int lane = tid & 31, wave = tid >> 5;
  const int mTile = wave & 3, ntBase = (wave >> 2) * 8;

  msg_gemm_layer<true, true>(bufA, bufB, packed + (size_t)(FRAG_MSG_L1 + var * 24) * 512,
                             b1s, wdts, mTile, ntBase, lane);
  __syncthreads();
  msg_gemm_layer<false, true>(bufB, bufA, packed + (size_t)(FRAG_MSG_L2 + var * 24) * 512,
                              b2s, nullptr, mTile, ntBase, lane);
  __syncthreads();

  // layer3: 64 -> 1, VALU dot (tiny); guard rows make all reads unconditional
  {
    const int pos = tid;
    float s = g3b[0];
#pragma unroll
    for (int k = 0; k < 3; ++k) {
      const int pp = pos + k - 1;
      for (int ch = 0; ch < 64; ++ch)
        s += w3s[ch * 3 + k] * (float)bufA[pp * LSTR + ch];
    }
    size_t oi;
    if (!tr) { const int row = (dir == 0) ? L : L + 1; oi = ((size_t)(dir * NB + b) * NH + row) * NW + pos; }
    else     { const int col = (dir == 2) ? L : L + 1; oi = ((size_t)(dir * NB + b) * NH + pos) * NW + col; }
    msg[oi] = s;
  }
}

// ---------------------------------------------------------------------------
// Aggregation layer1: 3x3 conv 5->64 via im2col (K = tap*5+c, padded to 64)
// Output NHWC bf16
// ---------------------------------------------------------------------------
__global__ __launch_bounds__(256) void agg_layer1(
    const float* __restrict__ msg, const float* __restrict__ dt,
    const bf16* __restrict__ afragBase, const float* __restrict__ bias,
    bf16* __restrict__ hout) {
  __shared__ bf16 xs[128 * LSTR];
  const int c0 = blockIdx.x * 128, r = blockIdx.y, b = blockIdx.z;
  const int tid = threadIdx.x;

  for (int i = tid; i < 48 * 128; i += 256) {
    const int kk = i >> 7, p = i & 127;
    float val = 0.0f;
    if (kk < 45) {
      const int tap = kk / 5, c = kk - tap * 5, dy = tap / 3, dx = tap - dy * 3;
      const int gr = r + dy - 1, gc = c0 + p + dx - 1;
      if (gr >= 0 && gr < NH && gc >= 0 && gc < NW) {
        if (c < 4) val = msg[((size_t)(c * NB + b) * NH + gr) * NW + gc];
        else       val = dt[((size_t)b * NH + gr) * NW + gc];
      }
    }
    xs[p * LSTR + kk] = (bf16)val;
  }
  __syncthreads();

  const int lane = tid & 31, wave = tid >> 5;
  const int n = lane & 15, hi16 = lane >> 4;
  const int mTile = wave & 3, ntBase = (wave >> 2) * 4;
  const v16bf A0 = load16(afragBase + (size_t)(0 * 4 + mTile) * 512 + lane * 16);
  const v16bf A1 = load16(afragBase + (size_t)(1 * 4 + mTile) * 512 + lane * 16);
  const int chBase = mTile * 16 + hi16 * 8;

  for (int q = 0; q < 4; ++q) {
    const int nt = ntBase + q;
    const bf16* srcp = &xs[(nt * 16 + n) * LSTR + hi16 * 16];
    v8f acc = zero8();
    acc = wmma_bf16(A0, load16(srcp), acc);
    acc = wmma_bf16(A1, load16(srcp + 32), acc);
    v8bf o;
#pragma unroll
    for (int v = 0; v < 8; ++v) {
      float x = acc[v] + bias[chBase + v];
      x = fmaxf(x, 0.0f);
      o[v] = (bf16)x;
    }
    const int col = c0 + nt * 16 + n;
    *(v8bf*)&hout[((size_t)(b * NH + r) * NW + col) * HID + chBase] = o;
  }
}

// ---------------------------------------------------------------------------
// Aggregation 3x3 conv, 64->COUT (shifted-GEMM over 9 taps, NHWC bf16 input)
//   COUT==64 -> bf16 NHWC out + ReLU ; COUT==1 -> f32 out (final layer)
// ---------------------------------------------------------------------------
template <int COUT, bool RELU>
__global__ __launch_bounds__(256) void agg_conv3x3(
    const bf16* __restrict__ hin, const bf16* __restrict__ afragBase,
    const float* __restrict__ bias, bf16* __restrict__ houtBf,
    float* __restrict__ houtF32) {
  __shared__ bf16 xs[3 * 130 * LSTR];
  const int c0 = blockIdx.x * 128, r = blockIdx.y, b = blockIdx.z;
  const int tid = threadIdx.x;

  for (int i = tid; i < 3 * 130 * 8; i += 256) {
    const int pix = i >> 3, ck = i & 7;
    const int d = pix / 130, p = pix - d * 130;
    const int gr = r + d - 1, gc = c0 + p - 1;
    v8bf val;
    if (gr >= 0 && gr < NH && gc >= 0 && gc < NW) {
      val = *(const v8bf*)&hin[((size_t)(b * NH + gr) * NW + gc) * HID + ck * 8];
    } else {
#pragma unroll
      for (int q = 0; q < 8; ++q) val[q] = (bf16)0.0f;
    }
    *(v8bf*)&xs[(d * 130 + p) * LSTR + ck * 8] = val;
  }
  __syncthreads();

  const int lane = tid & 31, wave = tid >> 5;
  const int n = lane & 15, hi16 = lane >> 4;
  constexpr int NTP = (COUT == 64) ? 4 : 1;
  const int mTile = (COUT == 64) ? (wave & 3) : 0;
  const int ntBase = (COUT == 64) ? (wave >> 2) * 4 : wave;

  v8f acc[NTP];
#pragma unroll
  for (int q = 0; q < NTP; ++q) acc[q] = zero8();

  for (int tap = 0; tap < 9; ++tap) {
    const int dy = tap / 3, dx = tap - dy * 3;
    v16bf A0, A1;
    if constexpr (COUT == 64) {
      A0 = load16(afragBase + (size_t)((tap * 2 + 0) * 4 + mTile) * 512 + lane * 16);
      A1 = load16(afragBase + (size_t)((tap * 2 + 1) * 4 + mTile) * 512 + lane * 16);
    } else {
      A0 = load16(afragBase + (size_t)(tap * 2 + 0) * 512 + lane * 16);
      A1 = load16(afragBase + (size_t)(tap * 2 + 1) * 512 + lane * 16);
    }
#pragma unroll
    for (int q = 0; q < NTP; ++q) {
      const int p = (ntBase + q) * 16 + n + dx;  // halo staged: always in [0,129]
      const bf16* srcp = &xs[(dy * 130 + p) * LSTR + hi16 * 16];
      acc[q] = wmma_bf16(A0, load16(srcp), acc[q]);
      acc[q] = wmma_bf16(A1, load16(srcp + 32), acc[q]);
    }
  }

  const int chBase = mTile * 16 + hi16 * 8;
#pragma unroll
  for (int q = 0; q < NTP; ++q) {
    const int col = c0 + (ntBase + q) * 16 + n;
    if constexpr (COUT == 64) {
      v8bf o;
#pragma unroll
      for (int v = 0; v < 8; ++v) {
        float x = acc[q][v] + bias[chBase + v];
        if constexpr (RELU) x = fmaxf(x, 0.0f);
        o[v] = (bf16)x;
      }
      *(v8bf*)&houtBf[((size_t)(b * NH + r) * NW + col) * HID + chBase] = o;
    } else {
      // M=0 lives in v=0 of lanes 0..15
      if (lane < 16) houtF32[(size_t)(b * NH + r) * NW + col] = acc[q][0] + bias[0];
    }
  }
}

// ---------------------------------------------------------------------------
// Host launcher
// ---------------------------------------------------------------------------
extern "C" void kernel_launch(void* const* d_in, const int* in_sizes, int n_in,
                              void* d_out, int out_size, void* d_ws, size_t ws_size,
                              hipStream_t stream) {
  const float* feat = (const float*)d_in[0];
  const float* dt   = (const float*)d_in[1];
  const float* g1w  = (const float*)d_in[2];
  const float* g1b  = (const float*)d_in[3];
  const float* g2w  = (const float*)d_in[4];
  const float* g2b  = (const float*)d_in[5];
  const float* g3w  = (const float*)d_in[6];
  const float* g3b  = (const float*)d_in[7];
  const float* a1w  = (const float*)d_in[8];
  const float* a1b  = (const float*)d_in[9];
  const float* a2w  = (const float*)d_in[10];
  const float* a2b  = (const float*)d_in[11];
  const float* a3w  = (const float*)d_in[12];
  const float* a3b  = (const float*)d_in[13];
  const int* iters  = (const int*)d_in[14];

  char* ws = (char*)d_ws;
  // Workspace layout (bytes):
  //   msg   : [4 dirs][B][H][W] f32                      @ 0         (4,194,304)
  //   packed: 194 A-fragments x 512 bf16                 @ 4,194,304 (198,656 -> pad)
  //   featT : [B][C][W][H] f32 (reused by h1 afterwards) @ 4,395,008 (33,554,432)
  //   dtT   : [B][W][H] f32                              @ 37,949,440(1,048,576)
  //   h1    : NHWC bf16 (aliases featT region)           @ 4,395,008 (33,554,432)
  //   h2    : NHWC bf16                                  @ 38,998,016(33,554,432)
  float* msg    = (float*)(ws + 0);
  bf16*  packed = (bf16*)(ws + 4194304);
  float* featT  = (float*)(ws + 4395008);
  float* dtT    = (float*)(ws + 37949440);
  bf16*  h1     = (bf16*)(ws + 4395008);
  bf16*  h2     = (bf16*)(ws + 38998016);
  float* outp   = (float*)d_out;

  hipMemsetAsync(msg, 0, (size_t)4 * NB * NH * NW * sizeof(float), stream);
  pack_weights<<<N_FRAGS, 32, 0, stream>>>(g1w, g2w, a1w, a2w, a3w, packed);
  transpose_k<<<dim3(NW / 32, NH / 32, NB * 33), dim3(32, 8), 0, stream>>>(feat, dt, featT, dtT);
  msg_kernel<<<dim3(NH - 1, NB, 4), 256, 0, stream>>>(feat, dt, featT, dtT,
                                                      g1w, g1b, g2b, g3w, g3b,
                                                      packed, msg, iters);
  agg_layer1<<<dim3(2, NH, NB), 256, 0, stream>>>(msg, dt, packed + (size_t)FRAG_AGG1 * 512, a1b, h1);
  agg_conv3x3<64, true><<<dim3(2, NH, NB), 256, 0, stream>>>(h1, packed + (size_t)FRAG_AGG2 * 512,
                                                             a2b, h2, nullptr);
  agg_conv3x3<1, false><<<dim3(2, NH, NB), 256, 0, stream>>>(h2, packed + (size_t)FRAG_AGG3 * 512,
                                                             a3b, nullptr, outp);
}
